// ConditionAttention_18614388261648
// MI455X (gfx1250) — matrix operand order repaired and checked by reading, not compiled, not run on hardware
//
#include <hip/hip_runtime.h>

#define NB    8
#define LSEQ  1024
#define EMBED 1024
#define HEADS 16
#define HDIM  64
#define SST   1040          // padded LDS stride (halves) for score rows
#define WOT_LD 40           // padded LDS stride (halves) for Wo tile

typedef _Float16 half_t;
typedef __attribute__((ext_vector_type(16))) _Float16 v16h;
typedef __attribute__((ext_vector_type(8)))  _Float16 v8h;
typedef __attribute__((ext_vector_type(8)))  float    v8f;

static __device__ __forceinline__ v8f wmma_f16(v16h a, v16h b, v8f c) {
  // D = A(16x32) * B(32x16) + C, f32 accumulate
  return __builtin_amdgcn_wmma_f32_16x16x32_f16(false, a, false, b, (short)0, c,
                                                false, false);
}

// A-fragment: 16x32 tile, row-major source, leading dim `ld` halves.
// CDNA5 layout: lane L holds row M=L&15; lanes 0-15 K={0..7,16..23}, lanes 16-31 K={8..15,24..31}.
static __device__ __forceinline__ v16h load_A_rm(const half_t* p, int ld, int lane) {
  int r  = lane & 15;
  int ko = (lane >> 4) << 3;
  const half_t* rp = p + r * ld;
  v8h lo = *(const v8h*)(rp + ko);
  v8h hi = *(const v8h*)(rp + 16 + ko);
  v16h a;
#pragma unroll
  for (int j = 0; j < 8; ++j) { a[j] = lo[j]; a[8 + j] = hi[j]; }
  return a;
}

// B-fragment: 32x16 tile where logical B[k][n] = p[n*ld + k] (stored transposed:
// K-matrix rows, pre-transposed V, Wo rows). Lane L holds column N=L&15;
// lanes 0-15 K={0..7,16..23}, lanes 16-31 K={8..15,24..31} (matches A layout).
static __device__ __forceinline__ v16h load_B_colrows(const half_t* p, int ld, int lane) {
  int nn = lane & 15;
  int ko = (lane >> 4) << 3;
  const half_t* rp = p + nn * ld + ko;
  v8h lo = *(const v8h*)(rp);
  v8h hi = *(const v8h*)(rp + 16);
  v16h b;
#pragma unroll
  for (int j = 0; j < 8; ++j) { b[j] = lo[j]; b[8 + j] = hi[j]; }
  return b;
}

// ---------------------------------------------------------------------------
// Phase 1: per-head projections -> f16 Q/K head-major [n][h][l][64],
// V transposed per head: [n][h][d][l] so attention V fragments are contiguous.
// ---------------------------------------------------------------------------
__global__ __launch_bounds__(256) void proj_kernel(
    const float* __restrict__ values, const float* __restrict__ keys,
    const float* __restrict__ query,  const float* __restrict__ position,
    const float* __restrict__ Wv, const float* __restrict__ Wk,
    const float* __restrict__ Wq, const float* __restrict__ Wpos,
    half_t* __restrict__ qh, half_t* __restrict__ kh, half_t* __restrict__ vt) {
  __shared__ float cvec[EMBED + 256];   // [query(1024) | pos_emb(256)]
  __shared__ float krow[EMBED];
  __shared__ float vrow[EMBED];
  int tok = blockIdx.x;                 // n*L + l
  int n = tok >> 10, l = tok & 1023;
  int tid = threadIdx.x;
  for (int i = tid; i < EMBED; i += 256) {
    cvec[i] = query [(size_t)tok * EMBED + i];
    krow[i] = keys  [(size_t)tok * EMBED + i];
    vrow[i] = values[(size_t)tok * EMBED + i];
  }
  // positional embedding: pos_emb[tid] = position[tok] . Wpos[tid]
  {
    float s = 0.f;
    const float* pr = position + (size_t)tok * 64;
    const float* wr = Wpos + tid * 64;
#pragma unroll 8
    for (int j = 0; j < 64; ++j) s += pr[j] * wr[j];
    cvec[EMBED + tid] = s;
  }
  __syncthreads();
#pragma unroll
  for (int i = 0; i < 4; ++i) {
    int c = tid + 256 * i;              // output channel 0..1023
    int h = c >> 6, d = c & 63;
    size_t oidx = ((((size_t)n * HEADS + h) * LSEQ) + l) * HDIM + d;
    size_t vidx = (((size_t)n * HEADS + h) * HDIM + d) * LSEQ + l;  // transposed
    float sk = 0.f, sv = 0.f;
    const float* wk = Wk + d * 64;
    const float* wv = Wv + d * 64;
    const float* kr = krow + h * 64;
    const float* vr = vrow + h * 64;
#pragma unroll 8
    for (int dd = 0; dd < 64; ++dd) { sk += kr[dd] * wk[dd]; sv += vr[dd] * wv[dd]; }
    kh[oidx] = (half_t)sk;
    vt[vidx] = (half_t)sv;
    // Q: head h consumes channels [80h, 80h+80) of the concatenated vector
    float sq = 0.f;
    const float* wq = Wq + d * 80;
    const float* qr = cvec + h * 80;
#pragma unroll 8
    for (int dd = 0; dd < 80; ++dd) sq += qr[dd] * wq[dd];
    qh[oidx] = (half_t)sq;
  }
}

__global__ __launch_bounds__(256) void convert_wo(const float* __restrict__ Wo,
                                                  half_t* __restrict__ woh) {
  int i = blockIdx.x * 256 + threadIdx.x;
  woh[i] = (half_t)Wo[i];
}

// ---------------------------------------------------------------------------
// Phase 2: attention. Block = (n, h, 64-query tile); 4 waves, one 16-row WMMA
// tile each. Full score rows (1024 keys) staged in LDS, f32 softmax.
// ---------------------------------------------------------------------------
__global__ __launch_bounds__(128) void attn_kernel(
    const half_t* __restrict__ qh, const half_t* __restrict__ kh,
    const half_t* __restrict__ vt, half_t* __restrict__ ctx) {
  extern __shared__ half_t S[];         // [64][SST] score scratch (130 KB)
  int bid = blockIdx.x;
  int qt = bid & 15;
  int h  = (bid >> 4) & 15;
  int n  = bid >> 8;
  int wave = threadIdx.x >> 5, lane = threadIdx.x & 31;
  const half_t* Q = qh + ((((size_t)n * HEADS + h) * LSEQ) + qt * 64) * HDIM;
  const half_t* K = kh + (((size_t)n * HEADS + h) * LSEQ) * HDIM;
  const half_t* V = vt + (((size_t)n * HEADS + h) * HDIM) * LSEQ;  // [d][l]
  half_t* Srow = S + wave * 16 * SST;
  int nn = lane & 15, rb = (lane >> 4) * 8;

  v16h aq0 = load_A_rm(Q + wave * 16 * HDIM,      HDIM, lane);
  v16h aq1 = load_A_rm(Q + wave * 16 * HDIM + 32, HDIM, lane);

  // S = (Q K^T) / sqrt(EMBED)
  for (int kt = 0; kt < LSEQ / 16; ++kt) {
    const half_t* Kt = K + kt * 16 * HDIM;
    __builtin_prefetch(Kt + 4 * 16 * HDIM, 0, 1);   // stream K ahead
    v16h b0 = load_B_colrows(Kt,      HDIM, lane);
    v16h b1 = load_B_colrows(Kt + 32, HDIM, lane);
    v8f c = {};
    c = wmma_f16(aq0, b0, c);
    c = wmma_f16(aq1, b1, c);
#pragma unroll
    for (int r = 0; r < 8; ++r)
      Srow[(rb + r) * SST + kt * 16 + nn] = (half_t)(c[r] * 0.03125f);
  }
  __syncthreads();

  // softmax: 2 threads per row, 128-bit LDS accesses, shfl combine
  {
    int r  = threadIdx.x >> 1;
    int hf = threadIdx.x & 1;
    half_t* row = S + r * SST + hf * (LSEQ / 2);
    float m = -1e30f;
    for (int i = 0; i < LSEQ / 2; i += 8) {
      v8h x = *(const v8h*)(row + i);
#pragma unroll
      for (int j = 0; j < 8; ++j) m = fmaxf(m, (float)x[j]);
    }
    m = fmaxf(m, __shfl_xor(m, 1, 32));
    float s = 0.f;
    for (int i = 0; i < LSEQ / 2; i += 8) {
      v8h x = *(const v8h*)(row + i);
      v8h e;
#pragma unroll
      for (int j = 0; j < 8; ++j) {
        float ev = __expf((float)x[j] - m);
        s += ev;
        e[j] = (half_t)ev;
      }
      *(v8h*)(row + i) = e;
    }
    s += __shfl_xor(s, 1, 32);
    float inv = 1.0f / s;
    for (int i = 0; i < LSEQ / 2; i += 8) {
      v8h x = *(const v8h*)(row + i);
      v8h o;
#pragma unroll
      for (int j = 0; j < 8; ++j) o[j] = (half_t)((float)x[j] * inv);
      *(v8h*)(row + i) = o;
    }
  }
  __syncthreads();

  // O = P V  (V transposed: B column n = V^T row n, contiguous)
  v8f acc[4] = {};
  for (int ks = 0; ks < LSEQ; ks += 32) {
    v16h ap = load_A_rm(Srow + ks, SST, lane);
#pragma unroll
    for (int t = 0; t < 4; ++t) {
      v16h bv = load_B_colrows(V + (t * 16) * LSEQ + ks, LSEQ, lane);
      acc[t] = wmma_f16(ap, bv, acc[t]);
    }
  }
  size_t rowg = (size_t)n * LSEQ + qt * 64 + wave * 16 + rb;
#pragma unroll
  for (int t = 0; t < 4; ++t)
#pragma unroll
    for (int r = 0; r < 8; ++r)
      ctx[(rowg + r) * EMBED + h * HDIM + t * 16 + nn] = (half_t)acc[t][r];
}

// ---------------------------------------------------------------------------
// Phase 3: out = ctx(8192x1024) @ Wo^T + bo. Block tile 128x64, Wo staged in LDS.
// ---------------------------------------------------------------------------
__global__ __launch_bounds__(256) void outproj_kernel(
    const half_t* __restrict__ ctx, const half_t* __restrict__ woh,
    const float* __restrict__ bo, float* __restrict__ out) {
  __shared__ half_t Wt[64 * WOT_LD];    // 64 rows x 32 K, padded stride
  int bn = blockIdx.x & 15;
  int bm = blockIdx.x >> 4;
  int wave = threadIdx.x >> 5, lane = threadIdx.x & 31;
  int tid = threadIdx.x;
  const half_t* A = ctx + ((size_t)bm * 128 + wave * 16) * EMBED;
  v8f acc[4] = {};
  for (int ks = 0; ks < EMBED; ks += 32) {
    {   // stage Wo[bn*64 .. +63][ks .. ks+31]
      int r  = tid >> 2;
      int c0 = (tid & 3) * 8;
      v8h v = *(const v8h*)(woh + (size_t)(bn * 64 + r) * EMBED + ks + c0);
      *(v8h*)(Wt + r * WOT_LD + c0) = v;
    }
    __syncthreads();
    v16h a = load_A_rm(A + ks, EMBED, lane);
#pragma unroll
    for (int t = 0; t < 4; ++t) {
      v16h b = load_B_colrows(Wt + (t * 16) * WOT_LD, WOT_LD, lane);
      acc[t] = wmma_f16(a, b, acc[t]);
    }
    __syncthreads();
  }
  int nn = lane & 15, rb = (lane >> 4) * 8;
#pragma unroll
  for (int t = 0; t < 4; ++t) {
    int col = bn * 64 + t * 16 + nn;
    float bias = bo[col];
#pragma unroll
    for (int r = 0; r < 8; ++r) {
      size_t rowi = (size_t)bm * 128 + wave * 16 + rb + r;
      out[rowi * EMBED + col] = acc[t][r] + bias;
    }
  }
}

extern "C" void kernel_launch(void* const* d_in, const int* in_sizes, int n_in,
                              void* d_out, int out_size, void* d_ws, size_t ws_size,
                              hipStream_t stream) {
  (void)in_sizes; (void)n_in; (void)out_size; (void)ws_size;
  const float* values   = (const float*)d_in[0];
  const float* keys     = (const float*)d_in[1];
  const float* query    = (const float*)d_in[2];
  const float* position = (const float*)d_in[3];
  const float* Wv   = (const float*)d_in[4];
  const float* Wk   = (const float*)d_in[5];
  const float* Wq   = (const float*)d_in[6];
  const float* Wpos = (const float*)d_in[7];
  const float* Wo   = (const float*)d_in[8];
  const float* bo   = (const float*)d_in[9];

  const size_t PH = (size_t)NB * HEADS * LSEQ * HDIM;   // 8,388,608 halves
  half_t* qh  = (half_t*)d_ws;
  half_t* kh  = qh + PH;
  half_t* vt  = kh + PH;                                // transposed V
  half_t* ctx = vt + PH;                                // [N*L][EMBED] halves
  half_t* woh = ctx + (size_t)NB * LSEQ * EMBED;        // [1024][1024] halves

  proj_kernel<<<NB * LSEQ, 256, 0, stream>>>(values, keys, query, position,
                                             Wv, Wk, Wq, Wpos, qh, kh, vt);
  convert_wo<<<(EMBED * EMBED) / 256, 256, 0, stream>>>(Wo, woh);
  attn_kernel<<<NB * HEADS * (LSEQ / 64), 128, 64 * SST * (int)sizeof(half_t),
                stream>>>(qh, kh, vt, ctx);
  outproj_kernel<<<(NB * LSEQ / 128) * (EMBED / 64), 256, 0, stream>>>(
      ctx, woh, bo, (float*)d_out);
}
